// PairwiseRankingLoss_60851096650048
// MI455X (gfx1250) — compile-verified, matrix operand order checked by reading, not run on hardware
//
#include <hip/hip_runtime.h>
#include <math.h>

// ---------------------------------------------------------------------------
// Contrastive (InfoNCE) loss, fused:
//   Xn = normalize(X), Yn = normalize(Y)
//   logits = Xn @ Yn^T / 0.07          (8192 x 8192, never materialized)
//   loss = mean_i( log(sum_j exp(logits_ij)) - logits_ii )
// Logits bounded by 1/T = 14.29 -> exp() is f32-safe, no max pass needed.
// GEMM: v_wmma_f32_16x16x32_f16, B tile staged in LDS via async global->LDS.
// ---------------------------------------------------------------------------

typedef __attribute__((ext_vector_type(16))) _Float16 v16h;
typedef __attribute__((ext_vector_type(8)))  _Float16 v8h;
typedef __attribute__((ext_vector_type(8)))  float    v8f;
typedef __attribute__((ext_vector_type(4)))  float    v4f;
typedef __attribute__((ext_vector_type(4)))  int      v4i;

#define D_DIM      256
#define INV_T      14.2857142857142857f   // 1 / 0.07
#define EPS_N      1e-8f
#define LDS_STRIDE 264                    // 256 + 8 f16 pad -> 528B row step
                                          // = 4-bank rotation, conflict-free

// ---- async global -> LDS copy of 16 bytes per lane ------------------------
__device__ __forceinline__ void async_copy_b128(_Float16* lds_dst,
                                                const _Float16* gsrc) {
#if __has_builtin(__builtin_amdgcn_global_load_async_to_lds_b128)
    // Builtin is typed over the 128-bit payload: (int4 as1*, int4 as3*, imm, imm)
    __builtin_amdgcn_global_load_async_to_lds_b128(
        (__attribute__((address_space(1))) v4i*)gsrc,
        (__attribute__((address_space(3))) v4i*)lds_dst,
        /*imm offset=*/0, /*cpol=*/0);
#else
    *(v8h*)lds_dst = *(const v8h*)gsrc;   // fallback: load + ds_store
#endif
}

__device__ __forceinline__ void wait_async_zero() {
#if __has_builtin(__builtin_amdgcn_s_wait_asynccnt)
    __builtin_amdgcn_s_wait_asynccnt(0);
#else
    asm volatile("s_wait_asynccnt 0x0" ::: "memory");
#endif
}

// ---------------------------------------------------------------------------
// Kernel 1: row-wise L2 normalize (f32 in) -> f16 out.  One wave per row.
// blockIdx.y selects X (0) or Y (1).
// ---------------------------------------------------------------------------
__global__ void __launch_bounds__(256)
nce_normalize_kernel(const float* __restrict__ X, const float* __restrict__ Y,
                     _Float16* __restrict__ Xh, _Float16* __restrict__ Yh,
                     int Brows)
{
    const float*    src = (blockIdx.y == 0) ? X  : Y;
    _Float16*       dst = (blockIdx.y == 0) ? Xh : Yh;

    const int wave = threadIdx.x >> 5;      // 8 waves per block
    const int lane = threadIdx.x & 31;
    const int row  = blockIdx.x * 8 + wave;
    if (row >= Brows) return;

    const float* rp = src + (size_t)row * D_DIM + lane * 8;
    v4f a = *(const v4f*)(rp);
    v4f b = *(const v4f*)(rp + 4);

    float ss = 0.f;
    #pragma unroll
    for (int i = 0; i < 4; ++i) ss += a[i] * a[i];
    #pragma unroll
    for (int i = 0; i < 4; ++i) ss += b[i] * b[i];

    #pragma unroll
    for (int m = 16; m >= 1; m >>= 1) ss += __shfl_xor(ss, m, 32);

    const float scale = 1.0f / fmaxf(sqrtf(ss), EPS_N);

    v8h o;
    #pragma unroll
    for (int i = 0; i < 4; ++i) o[i]     = (_Float16)(a[i] * scale);
    #pragma unroll
    for (int i = 0; i < 4; ++i) o[i + 4] = (_Float16)(b[i] * scale);
    *(v8h*)(dst + (size_t)row * D_DIM + lane * 8) = o;
}

// ---------------------------------------------------------------------------
// Kernel 2: WMMA GEMM tile + fused exp-row-sum + diagonal capture.
// Block = 256 threads = 8 waves; block tile = 128(M) x 128(N).
// Y tile (128 x 256 f16) staged ONCE per block into LDS (async), killing the
// 8x redundant per-wave B traffic; all 8 A k-fragments preloaded to VGPRs so
// the hot loop is ds_load_b128 + v_wmma only.
//
// Fragment addressing (ISA 7.12.2, wave32):
//   A 16x32 f16 : lane L -> row M = L&15; half (L>>4): K in
//                 {kh..kh+7} U {kh+16..kh+23}, kh = 8*(L>>4)
//   B 32x16 f16 : symmetric, lane L -> column N = L&15 (row N of Yn)
//   C/D 16x16 f32: VGPR v, lane L -> M = v + 8*(L>=16), N = L&15
// ---------------------------------------------------------------------------
__global__ void __launch_bounds__(256)
nce_gemm_softmax_kernel(const _Float16* __restrict__ Xh,
                        const _Float16* __restrict__ Yh,
                        float* __restrict__ rowsum,
                        float* __restrict__ diag,
                        int Brows)
{
    __shared__ _Float16 sY[128 * LDS_STRIDE];      // 67,584 B

    const int lane  = threadIdx.x & 31;
    const int wave  = threadIdx.x >> 5;            // 0..7
    const int m0    = blockIdx.y * 128 + wave * 16;
    const int n0    = blockIdx.x * 128;
    const int lrow  = lane & 15;
    const int khalf = (lane >> 4) << 3;            // 0 or 8

    // ---- stage Y[n0 : n0+128) x 256 into LDS (async, ASYNCcnt) ---------
    #pragma unroll
    for (int r = 0; r < 16; ++r) {
        const int row = r * 8 + wave;              // 0..127, unique
        async_copy_b128(&sY[(size_t)row * LDS_STRIDE + lane * 8],
                        Yh + (size_t)(n0 + row) * D_DIM + lane * 8);
    }
    wait_async_zero();
    __syncthreads();

    // ---- preload all A fragments (full K) for this wave's 16 rows ------
    union Frag { v16h v; v8h h[2]; };
    Frag af[8];
    const _Float16* aRow = Xh + (size_t)(m0 + lrow) * D_DIM + khalf;
    #pragma unroll
    for (int k = 0; k < 8; ++k) {
        af[k].h[0] = *(const v8h*)(aRow + k * 32);
        af[k].h[1] = *(const v8h*)(aRow + k * 32 + 16);
    }

    // ---- main loop: LDS B fragments + WMMA -----------------------------
    v8f c[8] = {};
    #pragma unroll
    for (int k = 0; k < 8; ++k) {
        #pragma unroll
        for (int t = 0; t < 8; ++t) {
            Frag bf;
            const _Float16* bp =
                &sY[(size_t)(t * 16 + lrow) * LDS_STRIDE + k * 32 + khalf];
            bf.h[0] = *(const v8h*)(bp);
            bf.h[1] = *(const v8h*)(bp + 16);

            c[t] = __builtin_amdgcn_wmma_f32_16x16x32_f16(
                /*neg_a=*/false, af[k].v, /*neg_b=*/false, bf.v,
                /*c_mod=*/(short)0, c[t], /*reuse_a=*/false, /*reuse_b=*/false);
        }
    }

    // ---- fused epilogue: exp-sum per row, diagonal capture -------------
    const int mhi = (lane >> 4) << 3;              // 0 or 8 (C layout)
    float psum[8];
    #pragma unroll
    for (int v = 0; v < 8; ++v) psum[v] = 0.0f;

    #pragma unroll
    for (int t = 0; t < 8; ++t) {
        const int j = n0 + t * 16 + lrow;          // global column
        #pragma unroll
        for (int v = 0; v < 8; ++v) {
            const int   i     = m0 + mhi + v;      // global row
            const float logit = c[t][v] * INV_T;
            psum[v] += __expf(logit);
            if (i == j) diag[i] = logit;           // hit exactly once in grid
        }
    }

    // reduce the 16 lanes sharing each row (xor within a 16-lane half)
    #pragma unroll
    for (int v = 0; v < 8; ++v) {
        float s = psum[v];
        s += __shfl_xor(s, 1, 32);
        s += __shfl_xor(s, 2, 32);
        s += __shfl_xor(s, 4, 32);
        s += __shfl_xor(s, 8, 32);
        if (lrow == 0) atomicAdd(&rowsum[m0 + mhi + v], s);
    }
}

// ---------------------------------------------------------------------------
// Kernel 3: loss = mean_i( log(rowsum[i]) - diag[i] ).  One block.
// ---------------------------------------------------------------------------
__global__ void __launch_bounds__(1024)
nce_loss_reduce_kernel(const float* __restrict__ rowsum,
                       const float* __restrict__ diag,
                       float* __restrict__ out, int Brows)
{
    __shared__ float sdata[1024];
    float acc = 0.0f;
    for (int i = threadIdx.x; i < Brows; i += blockDim.x)
        acc += logf(rowsum[i]) - diag[i];
    sdata[threadIdx.x] = acc;
    __syncthreads();
    for (int s = blockDim.x >> 1; s > 0; s >>= 1) {
        if (threadIdx.x < (unsigned)s) sdata[threadIdx.x] += sdata[threadIdx.x + s];
        __syncthreads();
    }
    if (threadIdx.x == 0) out[0] = sdata[0] / (float)Brows;
}

// ---------------------------------------------------------------------------
extern "C" void kernel_launch(void* const* d_in, const int* in_sizes, int n_in,
                              void* d_out, int out_size, void* d_ws, size_t ws_size,
                              hipStream_t stream)
{
    const float* X = (const float*)d_in[0];
    const float* Y = (const float*)d_in[1];
    float*     out = (float*)d_out;

    const int Brows = in_sizes[0] / D_DIM;         // 8192

    // workspace layout: Xh | Yh | rowsum | diag
    char*      ws   = (char*)d_ws;
    _Float16*  Xh   = (_Float16*)ws;
    _Float16*  Yh   = Xh + (size_t)Brows * D_DIM;
    float*   rowsum = (float*)(Yh + (size_t)Brows * D_DIM);
    float*     diag = rowsum + Brows;

    (void)hipMemsetAsync(rowsum, 0, (size_t)Brows * sizeof(float), stream);

    dim3 ngrid((Brows + 7) / 8, 2);                // y: 0 = X, 1 = Y
    nce_normalize_kernel<<<ngrid, 256, 0, stream>>>(X, Y, Xh, Yh, Brows);

    dim3 ggrid(Brows / 128, Brows / 128);          // 64 x 64 blocks
    nce_gemm_softmax_kernel<<<ggrid, 256, 0, stream>>>(Xh, Yh, rowsum, diag, Brows);

    nce_loss_reduce_kernel<<<1, 1024, 0, stream>>>(rowsum, diag, out, Brows);
}
